// AttentionFuse_29609504538851
// MI455X (gfx1250) — matrix operand order repaired and checked by reading, not compiled, not run on hardware
//
#include <hip/hip_runtime.h>
#include <cstdint>

#define DIM   1024
#define SEQ   2048
#define BATCH 8
#define NROWS (BATCH * SEQ)   // 16384

typedef __attribute__((ext_vector_type(16))) _Float16 v16h;
typedef __attribute__((ext_vector_type(8)))  float    v8f;

union Frag16 { v16h v; uint4 q[2]; _Float16 h[16]; };
union FragC  { v8f  v; float f[8]; };

// ---------------------------------------------------------------------------
// CDNA5 async global->LDS copy (ASYNCcnt path, no VGPR round trip).
// Generic pointers to LDS carry the LDS byte offset in their low 32 bits
// (flat-aperture rule), which is exactly the VDST operand of the instruction.
// ---------------------------------------------------------------------------
__device__ __forceinline__ void async_load_b128(const void* gptr, void* lptr) {
  const uint32_t lds = (uint32_t)(uintptr_t)lptr;
  asm volatile("global_load_async_to_lds_b128 %0, %1, off"
               :: "v"(lds), "v"(gptr) : "memory");
}
__device__ __forceinline__ void wait_async() {
  asm volatile("s_wait_asynccnt 0x0" ::: "memory");
}

// ---------------------------------------------------------------------------
// Kernel 1: transpose + convert weights: W[k][n] f32 -> WhT[n][k] f16
// ---------------------------------------------------------------------------
__global__ __launch_bounds__(256)
void pack_w_kernel(const float* __restrict__ Wq,
                   const float* __restrict__ Wk,
                   const float* __restrict__ Wv,
                   _Float16* __restrict__ WhT) {
  __shared__ float tile[32][33];
  const int w = blockIdx.z;
  const float* W = (w == 0) ? Wq : ((w == 1) ? Wk : Wv);
  _Float16* Wt = WhT + (size_t)w * DIM * DIM;
  const int k0 = blockIdx.x * 32;
  const int n0 = blockIdx.y * 32;
  const int tx = threadIdx.x;   // 0..31
  const int ty = threadIdx.y;   // 0..7
#pragma unroll
  for (int i = 0; i < 4; ++i)
    tile[ty + i * 8][tx] = W[(size_t)(k0 + ty + i * 8) * DIM + (n0 + tx)];
  __syncthreads();
#pragma unroll
  for (int i = 0; i < 4; ++i)
    Wt[(size_t)(n0 + ty + i * 8) * DIM + (k0 + tx)] =
        (_Float16)tile[tx][ty + i * 8];
}

// ---------------------------------------------------------------------------
// Kernel 2: Y = x @ W + b. 128x128 tile, BK=32, double-buffered LDS.
// B tile: async global->LDS. A tile: f32->f16 through registers (prefetched).
// ---------------------------------------------------------------------------
__global__ __launch_bounds__(256)
void qkv_gemm_kernel(const float* __restrict__ x,
                     const _Float16* __restrict__ Wt,
                     const float* __restrict__ bias,
                     _Float16* __restrict__ Y) {
  __shared__ __align__(16) _Float16 lA[2][128][32];
  __shared__ __align__(16) _Float16 lB[2][128][32];
  const int tid  = threadIdx.x;
  const int lane = tid & 31, wave = tid >> 5;
  const int lr = lane & 15, lh = lane >> 4;
  const int m0 = (wave & 3) * 32;
  const int n0 = (wave >> 2) * 64;
  const int mBase = blockIdx.x * 128;
  const int nBase = blockIdx.y * 128;

  const int ar = tid >> 3;          // A-stage row (0..31), 4 passes
  const int ac = (tid & 7) * 4;
  const int br = tid >> 2;          // B-stage row (0..63), 2 passes
  const int bc = (tid & 3) * 8;

  FragC acc[2][4];
  const v8f zero = {0.f, 0.f, 0.f, 0.f, 0.f, 0.f, 0.f, 0.f};
#pragma unroll
  for (int mi = 0; mi < 2; ++mi)
#pragma unroll
    for (int ni = 0; ni < 4; ++ni) acc[mi][ni].v = zero;

  // ---- prologue: stage k-block 0 into buffer 0 ----
#pragma unroll
  for (int p = 0; p < 2; ++p)
    async_load_b128(Wt + (size_t)(nBase + br + p * 64) * DIM + bc,
                    &lB[0][br + p * 64][bc]);
#pragma unroll
  for (int p = 0; p < 4; ++p) {
    const float4 f = *(const float4*)(x + (size_t)(mBase + ar + p * 32) * DIM + ac);
    union { _Float16 h[4]; uint2 u; } t;
    t.h[0] = (_Float16)f.x; t.h[1] = (_Float16)f.y;
    t.h[2] = (_Float16)f.z; t.h[3] = (_Float16)f.w;
    *(uint2*)&lA[0][ar + p * 32][ac] = t.u;
  }
  wait_async();
  __syncthreads();

  for (int k = 0; k < 32; ++k) {
    const int cur = k & 1, nxt = cur ^ 1;
    const int kb  = (k + 1) * 32;
    float4 fpref[4];
    if (k < 31) {
      // issue next B tile (async -> LDS) and next A tile (global -> regs)
#pragma unroll
      for (int p = 0; p < 2; ++p)
        async_load_b128(Wt + (size_t)(nBase + br + p * 64) * DIM + kb + bc,
                        &lB[nxt][br + p * 64][bc]);
#pragma unroll
      for (int p = 0; p < 4; ++p)
        fpref[p] = *(const float4*)(x + (size_t)(mBase + ar + p * 32) * DIM + kb + ac);
    }

    // ---- compute on current buffer ----
    Frag16 a[2], b[4];
#pragma unroll
    for (int mi = 0; mi < 2; ++mi) {
      const int r = m0 + mi * 16 + lr;
      a[mi].q[0] = *(const uint4*)&lA[cur][r][8 * lh];
      a[mi].q[1] = *(const uint4*)&lA[cur][r][16 + 8 * lh];
    }
#pragma unroll
    for (int ni = 0; ni < 4; ++ni) {
      const int n = n0 + ni * 16 + lr;
      b[ni].q[0] = *(const uint4*)&lB[cur][n][8 * lh];
      b[ni].q[1] = *(const uint4*)&lB[cur][n][16 + 8 * lh];
    }
#pragma unroll
    for (int mi = 0; mi < 2; ++mi)
#pragma unroll
      for (int ni = 0; ni < 4; ++ni)
        acc[mi][ni].v = __builtin_amdgcn_wmma_f32_16x16x32_f16(
            false, a[mi].v, false, b[ni].v, (short)0, acc[mi][ni].v,
            false, false);

    if (k < 31) {
      // convert + store prefetched A into the alternate buffer
#pragma unroll
      for (int p = 0; p < 4; ++p) {
        union { _Float16 h[4]; uint2 u; } t;
        t.h[0] = (_Float16)fpref[p].x; t.h[1] = (_Float16)fpref[p].y;
        t.h[2] = (_Float16)fpref[p].z; t.h[3] = (_Float16)fpref[p].w;
        *(uint2*)&lA[nxt][ar + p * 32][ac] = t.u;
      }
      wait_async();
    }
    __syncthreads();
  }

#pragma unroll
  for (int ni = 0; ni < 4; ++ni) {
    const int col = nBase + n0 + ni * 16 + lr;
    const float bv = bias[col];
#pragma unroll
    for (int mi = 0; mi < 2; ++mi)
#pragma unroll
      for (int j = 0; j < 8; ++j) {
        const int row = mBase + m0 + mi * 16 + 8 * lh + j;
        Y[(size_t)row * DIM + col] = (_Float16)(acc[mi][ni].f[j] + bv);
      }
  }
}

// ---------------------------------------------------------------------------
// Kernel 3: P[b][q][t] = (Q[b,q,:] . K[b,t,:]) / sqrt(D), f16 out.
// Both tiles are f16 in global -> fully async double-buffered staging.
// ---------------------------------------------------------------------------
__global__ __launch_bounds__(256)
void scores_gemm_kernel(const _Float16* __restrict__ Qh,
                        const _Float16* __restrict__ Kh,
                        _Float16* __restrict__ P) {
  __shared__ __align__(16) _Float16 lA[2][128][32];
  __shared__ __align__(16) _Float16 lB[2][128][32];
  const int tid  = threadIdx.x;
  const int lane = tid & 31, wave = tid >> 5;
  const int lr = lane & 15, lh = lane >> 4;
  const int m0 = (wave & 3) * 32;
  const int n0 = (wave >> 2) * 64;
  const int qBase  = blockIdx.x * 128;
  const int kvBase = blockIdx.y * 128;
  const int bb = blockIdx.z;
  const _Float16* Qb = Qh + (size_t)bb * SEQ * DIM;
  const _Float16* Kb = Kh + (size_t)bb * SEQ * DIM;
  _Float16* Pb = P + (size_t)bb * SEQ * SEQ;

  const int sr = tid >> 2;          // stage row (0..63), 2 passes
  const int sc = (tid & 3) * 8;

  FragC acc[2][4];
  const v8f zero = {0.f, 0.f, 0.f, 0.f, 0.f, 0.f, 0.f, 0.f};
#pragma unroll
  for (int mi = 0; mi < 2; ++mi)
#pragma unroll
    for (int ni = 0; ni < 4; ++ni) acc[mi][ni].v = zero;

  // ---- prologue ----
#pragma unroll
  for (int p = 0; p < 2; ++p) {
    async_load_b128(Qb + (size_t)(qBase + sr + p * 64) * DIM + sc,
                    &lA[0][sr + p * 64][sc]);
    async_load_b128(Kb + (size_t)(kvBase + sr + p * 64) * DIM + sc,
                    &lB[0][sr + p * 64][sc]);
  }
  wait_async();
  __syncthreads();

  for (int k = 0; k < 32; ++k) {
    const int cur = k & 1, nxt = cur ^ 1;
    const int kb  = (k + 1) * 32;
    if (k < 31) {
#pragma unroll
      for (int p = 0; p < 2; ++p) {
        async_load_b128(Qb + (size_t)(qBase + sr + p * 64) * DIM + kb + sc,
                        &lA[nxt][sr + p * 64][sc]);
        async_load_b128(Kb + (size_t)(kvBase + sr + p * 64) * DIM + kb + sc,
                        &lB[nxt][sr + p * 64][sc]);
      }
    }

    Frag16 a[2], b[4];
#pragma unroll
    for (int mi = 0; mi < 2; ++mi) {
      const int rr = m0 + mi * 16 + lr;
      a[mi].q[0] = *(const uint4*)&lA[cur][rr][8 * lh];
      a[mi].q[1] = *(const uint4*)&lA[cur][rr][16 + 8 * lh];
    }
#pragma unroll
    for (int ni = 0; ni < 4; ++ni) {
      const int n = n0 + ni * 16 + lr;
      b[ni].q[0] = *(const uint4*)&lB[cur][n][8 * lh];
      b[ni].q[1] = *(const uint4*)&lB[cur][n][16 + 8 * lh];
    }
#pragma unroll
    for (int mi = 0; mi < 2; ++mi)
#pragma unroll
      for (int ni = 0; ni < 4; ++ni)
        acc[mi][ni].v = __builtin_amdgcn_wmma_f32_16x16x32_f16(
            false, a[mi].v, false, b[ni].v, (short)0, acc[mi][ni].v,
            false, false);

    if (k < 31) wait_async();
    __syncthreads();
  }

  const float scale = 0.03125f;   // 1/sqrt(1024)
#pragma unroll
  for (int ni = 0; ni < 4; ++ni) {
    const int col = kvBase + n0 + ni * 16 + lr;
#pragma unroll
    for (int mi = 0; mi < 2; ++mi)
#pragma unroll
      for (int j = 0; j < 8; ++j) {
        const int row = qBase + m0 + mi * 16 + 8 * lh + j;
        Pb[(size_t)row * SEQ + col] = (_Float16)(acc[mi][ni].f[j] * scale);
      }
  }
}

// ---------------------------------------------------------------------------
// Kernel 4: row softmax over P (in place), 256 threads per 2048-wide row.
// ---------------------------------------------------------------------------
__device__ __forceinline__ float waveMax(float v) {
#pragma unroll
  for (int o = 16; o > 0; o >>= 1) v = fmaxf(v, __shfl_down(v, o, 32));
  return v;
}
__device__ __forceinline__ float waveSum(float v) {
#pragma unroll
  for (int o = 16; o > 0; o >>= 1) v += __shfl_down(v, o, 32);
  return v;
}

__global__ __launch_bounds__(256)
void softmax_kernel(_Float16* __restrict__ P) {
  __shared__ float redA[8], redB[8];
  _Float16* p = P + (size_t)blockIdx.x * SEQ;
  const int tid = threadIdx.x;
  union { uint4 q; _Float16 h[8]; } d;
  d.q = *(const uint4*)(p + tid * 8);
  float v[8];
  float m = -3.0e38f;
#pragma unroll
  for (int i = 0; i < 8; ++i) { v[i] = (float)d.h[i]; m = fmaxf(m, v[i]); }
  m = waveMax(m);
  if ((tid & 31) == 0) redA[tid >> 5] = m;
  __syncthreads();
  m = redA[0];
#pragma unroll
  for (int i = 1; i < 8; ++i) m = fmaxf(m, redA[i]);
  float s = 0.f;
#pragma unroll
  for (int i = 0; i < 8; ++i) { v[i] = __expf(v[i] - m); s += v[i]; }
  s = waveSum(s);
  if ((tid & 31) == 0) redB[tid >> 5] = s;
  __syncthreads();
  s = redB[0];
#pragma unroll
  for (int i = 1; i < 8; ++i) s += redB[i];
  const float inv = 1.0f / s;
#pragma unroll
  for (int i = 0; i < 8; ++i) d.h[i] = (_Float16)(v[i] * inv);
  *(uint4*)(p + tid * 8) = d.q;
}

// ---------------------------------------------------------------------------
// Kernel 5: abar[b][t] = mean_s P[b][s][t]
// ---------------------------------------------------------------------------
__global__ __launch_bounds__(256)
void colmean_kernel(const _Float16* __restrict__ P, float* __restrict__ abar) {
  const int bb = blockIdx.y;
  const int t  = blockIdx.x * 256 + threadIdx.x;
  const _Float16* p = P + (size_t)bb * SEQ * SEQ + t;
  float acc = 0.f;
#pragma unroll 8
  for (int s = 0; s < SEQ; ++s) acc += (float)p[(size_t)s * SEQ];
  abar[bb * SEQ + t] = acc * (1.0f / (float)SEQ);
}

// ---------------------------------------------------------------------------
// Kernel 6: out[b][d] = sum_t abar[b][t] * V[b][t][d]   (exact mean fusion)
// ---------------------------------------------------------------------------
__global__ __launch_bounds__(256)
void final_kernel(const float* __restrict__ abar,
                  const _Float16* __restrict__ Vh,
                  float* __restrict__ out) {
  const int bb = blockIdx.y;
  const int d  = blockIdx.x * 256 + threadIdx.x;
  const _Float16* v = Vh + (size_t)bb * SEQ * DIM + d;
  const float* a = abar + bb * SEQ;
  float acc = 0.f;
#pragma unroll 8
  for (int t = 0; t < SEQ; ++t) acc += a[t] * (float)v[(size_t)t * DIM];
  out[bb * DIM + d] = acc;
}

// ---------------------------------------------------------------------------
// Workspace layout (bytes)
// ---------------------------------------------------------------------------
static const size_t OFF_Q = 0;                       // 32 MiB f16 Q
static const size_t OFF_K = 33554432;                // 32 MiB f16 K
static const size_t OFF_V = 67108864;                // 32 MiB f16 V
static const size_t OFF_W = 100663296;               //  6 MiB f16 WhT x3
static const size_t OFF_P = 106954752;               // 64 MiB f16 scores/attn
static const size_t OFF_A = 174063616;               // 64 KiB f32 abar

extern "C" void kernel_launch(void* const* d_in, const int* in_sizes, int n_in,
                              void* d_out, int out_size, void* d_ws,
                              size_t ws_size, hipStream_t stream) {
  const float* x  = (const float*)d_in[0];
  const float* Wq = (const float*)d_in[1];
  const float* bq = (const float*)d_in[2];
  const float* Wk = (const float*)d_in[3];
  const float* bk = (const float*)d_in[4];
  const float* Wv = (const float*)d_in[5];
  const float* bv = (const float*)d_in[6];
  char* ws = (char*)d_ws;

  _Float16* Qh  = (_Float16*)(ws + OFF_Q);
  _Float16* Kh  = (_Float16*)(ws + OFF_K);
  _Float16* Vh  = (_Float16*)(ws + OFF_V);
  _Float16* WhT = (_Float16*)(ws + OFF_W);
  _Float16* P   = (_Float16*)(ws + OFF_P);
  float*    ab  = (float*)(ws + OFF_A);
  float*    out = (float*)d_out;

  // 1) pack weights (transpose + f32->f16)
  pack_w_kernel<<<dim3(32, 32, 3), dim3(32, 8), 0, stream>>>(Wq, Wk, Wv, WhT);

  // 2) Q/K/V projections (WMMA, async double-buffered)
  qkv_gemm_kernel<<<dim3(128, 8), 256, 0, stream>>>(
      x, WhT + (size_t)0 * DIM * DIM, bq, Qh);
  qkv_gemm_kernel<<<dim3(128, 8), 256, 0, stream>>>(
      x, WhT + (size_t)1 * DIM * DIM, bk, Kh);
  qkv_gemm_kernel<<<dim3(128, 8), 256, 0, stream>>>(
      x, WhT + (size_t)2 * DIM * DIM, bv, Vh);

  // 3) scores = scale * Q K^T (WMMA, async double-buffered)
  scores_gemm_kernel<<<dim3(16, 16, 8), 256, 0, stream>>>(Qh, Kh, P);

  // 4) softmax rows
  softmax_kernel<<<dim3(NROWS), 256, 0, stream>>>(P);

  // 5) column-mean of attention
  colmean_kernel<<<dim3(8, 8), 256, 0, stream>>>(P, ab);

  // 6) out = abar @ V
  final_kernel<<<dim3(4, 8), 256, 0, stream>>>(ab, Vh, out);
}
